// MultiScaleImageFilters_81879256531347
// MI455X (gfx1250) — compile-verified, matrix operand order checked by reading, not run on hardware
//
#include <hip/hip_runtime.h>
#include <hip/hip_bf16.h>

typedef __attribute__((ext_vector_type(16))) _Float16 v16h;
typedef __attribute__((ext_vector_type(8)))  float    v8f;

#define IMG_W   512
#define IMG_H   512
#define PLANES  48            // 16 batch * 3 channels
#define PLANE_SZ (IMG_W * IMG_H)
#define TENSOR_SZ (PLANES * PLANE_SZ)   // 12,582,912 elements per output tensor

#define TILE_ROWS 32          // output rows per workgroup
#define LDS_W 520             // 3 left halo + 512 + 4 right halo (519) padded to 520
#define LDS_H 39              // 3 top halo + 32 + 4 bottom halo

#define PI_F 3.14159265358979323846f

// gfx1250 has a hardware transcendental V_TANH_F32 (ISA: TRANS op list).
__device__ __forceinline__ float fast_tanh(float x) {
#if __has_builtin(__builtin_amdgcn_tanhf)
    return __builtin_amdgcn_tanhf(x);
#elif __has_builtin(__builtin_amdgcn_tanh_f32)
    return __builtin_amdgcn_tanh_f32(x);
#else
    float r;
    asm("v_tanh_f32 %0, %1" : "=v"(r) : "v"(x));
    return r;
#endif
}

// Unnormalized filter weight for filter m at 7x7 tap (dy,dx), center (3,3).
// m: 0=lpf(3x3/16), 1..3=LoG r=1,2,3 (sigma .8/1.0/1.4), 4=hpf3, 5=hpf5,
//    6..9=gabor 0/45/90/135 deg.
__device__ __forceinline__ float unnorm_weight(int m, int dy, int dx) {
    int ry = dy - 3, rx = dx - 3;
    int ay = ry < 0 ? -ry : ry, ax = rx < 0 ? -rx : rx;
    switch (m) {
    case 0: {
        if (ay > 1 || ax > 1) return 0.f;
        float ty = (ry == 0) ? 2.f : 1.f;
        float tx = (rx == 0) ? 2.f : 1.f;
        return ty * tx * (1.f / 16.f);
    }
    case 1: case 2: case 3: {
        int rad = m;
        float sigma = (m == 1) ? 0.8f : ((m == 2) ? 1.0f : 1.4f);
        if (ay > rad || ax > rad) return 0.f;
        float r2 = (float)(ry * ry + rx * rx);
        float s2 = sigma * sigma;
        float g   = expf(-r2 / (2.f * s2));
        float lap = -1.f / (PI_F * s2 * s2) * (1.f - r2 / (2.f * s2));
        return lap * g;
    }
    case 4: {
        if (ay > 1 || ax > 1) return 0.f;
        return (ry == 0 && rx == 0) ? 8.f : -1.f;
    }
    case 5: {
        if (ay > 2 || ax > 2) return 0.f;
        return (ry == 0 && rx == 0) ? 24.f : -1.f;
    }
    case 6: case 7: case 8: case 9: {
        float th = (float)(m - 6) * 45.f * PI_F / 180.f;
        float x = (float)rx, y = (float)ry;           // meshgrid: x along cols, y along rows
        float ct = cosf(th), st = sinf(th);
        float xt =  x * ct + y * st;
        float yt = -x * st + y * ct;
        float g = expf(-(xt * xt / 4.f + yt * yt / 4.f) * 0.5f) *
                  cosf(2.f * PI_F * 0.1f * xt);
        return g;
    }
    default: return 0.f;
    }
}

// One 16x16x32 f16 WMMA K-slice element index for lane half `hs` and v16h slot j.
// CDNA5 16-bit A-matrix layout: VGPR0..3 -> K 0..7 (lanes 0-15) / 8..15
// (lanes 16-31); VGPR4..7 -> K 16..23 / 24..31. With 8-wide tap rows this makes
// each lane's 8-slot group contiguous in LDS (compiler emits ds_load_b128).
__device__ __forceinline__ int slot_to_k(int j, int hs) {
    int vg = j >> 1, hi = j & 1;
    return ((vg >> 2) << 4) + (hs << 3) + ((vg & 3) << 1) + hi;
}

__global__ __launch_bounds__(256)
void MultiScaleImageFilters_81879256531347_kernel(const float* __restrict__ xin,
                                                  float* __restrict__ out,
                                                  int nout) {
    __shared__ _Float16 tile[LDS_H * LDS_W];   // ~40.6 KB of 320 KB WGP LDS

    const int plane    = blockIdx.x >> 4;          // 48 planes
    const int tile_row = blockIdx.x & 15;          // 16 row-tiles per plane
    const int gy0      = tile_row * TILE_ROWS;
    const long plane_off = (long)plane * PLANE_SZ;

    // ---- Stage 1: fill LDS f16 tile (with zero-padded 7x7 halo) ----
    for (int i = threadIdx.x; i < LDS_H * LDS_W; i += 256) {
        int ly = i / LDS_W, lx = i - ly * LDS_W;
        int gy = gy0 - 3 + ly;
        int gx = lx - 3;
        float v = 0.f;
        if (gy >= 0 && gy < IMG_H && gx >= 0 && gx < IMG_W && lx < 519)
            v = xin[plane_off + (long)gy * IMG_W + gx];
        tile[i] = (_Float16)v;
    }
    __syncthreads();

    const int lane    = threadIdx.x & 31;
    const int wave    = threadIdx.x >> 5;
    const int n       = lane & 15;        // pixel column inside 16-wide strip
    const int halfsel = lane >> 4;        // which K-half this lane holds

    // ---- Stage 2: build A (weights) in registers, per ISA A-matrix layout ----
    const int m = lane & 15;              // filter row for this lane
    float norm = 1.f;
    if ((m >= 1 && m <= 3) || (m >= 6 && m <= 9)) {
        norm = 0.f;
        for (int dy = 0; dy < 7; ++dy)
            for (int dx = 0; dx < 7; ++dx)
                norm += fabsf(unnorm_weight(m, dy, dx));
    }
    v16h a0 = {}, a1 = {};
    if (m < 10) {
        #pragma unroll
        for (int j = 0; j < 16; ++j) {
            int k = slot_to_k(j, halfsel);
            {   // K-block 0: taps 0..31  (tap K: dy = K>>3, dx = K&7; 8-wide rows)
                int dy = k >> 3, dx = k & 7;
                float w = (dx < 7) ? unnorm_weight(m, dy, dx) / norm : 0.f;
                a0[j] = (_Float16)w;
            }
            {   // K-block 1: taps 32..63
                int K = k + 32;
                int dy = K >> 3, dx = K & 7;
                float w = (dy < 7 && dx < 7) ? unnorm_weight(m, dy, dx) / norm : 0.f;
                a1[j] = (_Float16)w;
            }
        }
    }

    // ---- Stage 3: WMMA over 16-pixel strips; 2 matrix ops cover all 10 filters ----
    for (int rr = 0; rr < TILE_ROWS / 8; ++rr) {
        const int r_t = wave * (TILE_ROWS / 8) + rr;   // local output row 0..31
        const int gy  = gy0 + r_t;
        for (int tx = 0; tx < IMG_W / 16; ++tx) {
            const int colbase = tx * 16 + n;           // LDS col of tap dx=0

            v16h b0, b1;                               // im2col column for pixel n
            #pragma unroll
            for (int j = 0; j < 16; ++j) {
                int k = slot_to_k(j, halfsel);
                {
                    int dy = k >> 3, dx = k & 7;
                    b0[j] = tile[(r_t + dy) * LDS_W + colbase + dx];
                }
                {
                    int K = k + 32;
                    int dy = K >> 3, dx = K & 7;
                    b1[j] = tile[(r_t + dy) * LDS_W + colbase + dx];
                }
            }

            v8f acc = {};
            acc = __builtin_amdgcn_wmma_f32_16x16x32_f16(
                      false, a0, false, b0, (short)0, acc, false, false);
            acc = __builtin_amdgcn_wmma_f32_16x16x32_f16(
                      false, a1, false, b1, (short)0, acc, false, false);

            // Elementwise tanh(0.1*conv) on the raw D tile: lanes<16 cover
            // filters 0..7, lanes>=16 cover filters 8..15 (8,9 = gabor 90/135).
            float t[8];
            #pragma unroll
            for (int r = 0; r < 8; ++r)
                t[r] = fast_tanh(acc[r] * 0.1f);

            // Bring tanh(filter8/9) down to the storing lanes (whole wave active).
            float g90  = __shfl_xor(t[0], 16, 32);
            float g135 = __shfl_xor(t[1], 16, 32);

            if (halfsel == 0) {
                // squash is a monotonic affine of tanh => max commutes.
                float o_lpf = acc[0];
                float o_log = (fmaxf(fmaxf(t[1], t[2]), t[3]) + 1.f) * 0.5f;
                float o_hpf = (fmaxf(t[4], t[5]) + 1.f) * 0.5f;
                float o_gab = fmaxf(fmaxf(t[6], t[7]), fmaxf(g90, g135));

                long idx = plane_off + (long)gy * IMG_W + (tx * 16 + n);
                out[idx]                        = xin[idx];   // 'original' (exact copy)
                out[(long)TENSOR_SZ     + idx]  = o_lpf;
                out[(long)TENSOR_SZ * 2 + idx]  = o_log;
                out[(long)TENSOR_SZ * 3 + idx]  = o_hpf;
                if (nout >= 5)
                    out[(long)TENSOR_SZ * 4 + idx] = o_gab;
            }
        }
    }
}

extern "C" void kernel_launch(void* const* d_in, const int* in_sizes, int n_in,
                              void* d_out, int out_size, void* d_ws, size_t ws_size,
                              hipStream_t stream) {
    (void)in_sizes; (void)n_in; (void)d_ws; (void)ws_size;
    const float* xin = (const float*)d_in[0];
    float* out = (float*)d_out;
    int nout = out_size / TENSOR_SZ;   // 5 when use_gabor, else 4
    dim3 grid(PLANES * (IMG_H / TILE_ROWS));   // 48 * 16 = 768 workgroups
    dim3 block(256);                           // 8 wave32s
    MultiScaleImageFilters_81879256531347_kernel<<<grid, block, 0, stream>>>(xin, out, nout);
}
